// ConsistencyDetector_21835613733616
// MI455X (gfx1250) — matrix operand order repaired and checked by reading, not compiled, not run on hardware
//
#include <hip/hip_runtime.h>
#include <hip/hip_bf16.h>

// CDNA5 / gfx1250 WMMA types (wave32)
typedef __attribute__((ext_vector_type(16))) __bf16 v16bf;
typedef __attribute__((ext_vector_type(8)))  __bf16 v8bf;
typedef __attribute__((ext_vector_type(8)))  float  v8f;
typedef __attribute__((ext_vector_type(4)))  float  f4v;

#define FDIM 64      // F == H == 64
#define DECD 128

// ---------------------------------------------------------------------------
// fp32 atomic add -> global_atomic_add_f32 (HW float atomic, no CAS loop)
// ---------------------------------------------------------------------------
__device__ __forceinline__ void atomAddF32(float* p, float v) {
  unsafeAtomicAdd(p, v);
}

// ---------------------------------------------------------------------------
// init: deg = 1.0 (self-loop), zero the two scatter accumulators
// ---------------------------------------------------------------------------
__global__ void init_kernel(float* __restrict__ deg, float* __restrict__ macc,
                            float* __restrict__ hmacc, int N) {
  int i = blockIdx.x * blockDim.x + threadIdx.x;
  if (i < N) deg[i] = 1.0f;
  int NF = N * FDIM;
  if (i < NF) { macc[i] = 0.0f; hmacc[i] = 0.0f; }
}

// ---------------------------------------------------------------------------
// degree: deg[dst[e]] += 1  (float counts exact well past max degree)
// ---------------------------------------------------------------------------
__global__ void degree_kernel(const int* __restrict__ dst, float* __restrict__ deg, int E) {
  int e = blockIdx.x * blockDim.x + threadIdx.x;
  if (e < E) atomAddF32(&deg[dst[e]], 1.0f);
}

// ---------------------------------------------------------------------------
// scatter: acc[dst[e], f] += feat[src[e], f]   (64 feature lanes per edge)
// ---------------------------------------------------------------------------
__global__ void scatter_add_kernel(const float* __restrict__ feat,
                                   const int* __restrict__ src,
                                   const int* __restrict__ dst,
                                   float* __restrict__ acc, int E) {
  long long t = (long long)blockIdx.x * blockDim.x + threadIdx.x;
  long long e = t >> 6;            // FDIM == 64 features per edge
  int f = (int)(t & 63);
  if (e >= E) return;
  int s = src[e];
  int d = dst[e];
  atomAddF32(&acc[(size_t)d * FDIM + f], feat[(size_t)s * FDIM + f]);
}

// ---------------------------------------------------------------------------
// finish mean: acc = (acc + self) * (1/deg)    [self-loop + normalization]
// ---------------------------------------------------------------------------
__global__ void finish_mean_kernel(const float* __restrict__ self,
                                   float* __restrict__ acc,
                                   const float* __restrict__ deg, int N) {
  long long t = (long long)blockIdx.x * blockDim.x + threadIdx.x;
  long long v = t >> 6;
  int f = (int)(t & 63);
  if (v >= N) return;
  float inv = 1.0f / fmaxf(deg[v], 1.0f);
  size_t idx = (size_t)v * FDIM + f;
  acc[idx] = (acc[idx] + self[idx]) * inv;
}

// ---------------------------------------------------------------------------
// WMMA GEMM: Out[M x Nc] = act(A[M x K] @ W[K x Nc] + bias), bf16 / f32 acc.
//
// - W staged once per block into LDS, bf16, transposed [col][k] with +8-elem
//   row padding: B fragments = two ds_load_b128 per lane, bank-conflict-free.
// - One wave32 per 16-row stripe computes the full 16 x Nc output (NT = Nc/16
//   register-blocked accumulators) -> A fragment loaded once per K-step via
//   aligned float4 global loads, 8..16 v_wmma per fully-unrolled body.
// - Stripe id is wave-uniform; the uniform exit is after the barrier, so all
//   WMMAs run with EXEC all-1s (ISA §7.12 requirement).
// ---------------------------------------------------------------------------
template <int K, int NT>
__global__ void __launch_bounds__(256)
gemm_bias_act_kernel(const float* __restrict__ A, const float* __restrict__ W,
                     const float* __restrict__ bias, float* __restrict__ Out,
                     int M, int relu) {
  constexpr int Nc = NT * 16;
  constexpr int KP = K + 8;                    // padded LDS row (bf16 elems)
  __shared__ __bf16 Wl[Nc * KP];

  // cooperative coalesced load of W (row-major [K][Nc]) -> LDS transposed
  for (int idx = threadIdx.x; idx < K * Nc; idx += blockDim.x) {
    int k = idx / Nc;
    int c = idx - k * Nc;
    Wl[c * KP + k] = (__bf16)W[idx];
  }
  __syncthreads();

  const int stripe = blockIdx.x * (blockDim.x >> 5) + (int)(threadIdx.x >> 5);
  const int nStripes = M >> 4;                 // M is a multiple of 16
  if (stripe >= nStripes) return;              // wave-uniform exit

  const int lane = (int)(threadIdx.x & 31);
  const int r    = lane & 15;                  // A row in tile / B,C column
  const int hi   = lane >> 4;
  const int kbA  = hi ? 8 : 0;                 // A half-wave K split (ISA layout)
  const int kbB  = hi ? 8 : 0;                 // B: same K split as A (table order)
  const int m0   = stripe << 4;

  const float* Arow = A + (size_t)(m0 + r) * K;

  v8f acc[NT] = {};

#pragma unroll
  for (int k0 = 0; k0 < K; k0 += 32) {
    // A fragment: a[0..7] = K kbA..kbA+7, a[8..15] = K kbA+16..kbA+23
    const f4v f0 = *(const f4v*)(Arow + k0 + kbA);
    const f4v f1 = *(const f4v*)(Arow + k0 + kbA + 4);
    const f4v f2 = *(const f4v*)(Arow + k0 + kbA + 16);
    const f4v f3 = *(const f4v*)(Arow + k0 + kbA + 20);
    v16bf a;
#pragma unroll
    for (int i = 0; i < 4; ++i) {
      a[i]      = (__bf16)f0[i];
      a[4 + i]  = (__bf16)f1[i];
      a[8 + i]  = (__bf16)f2[i];
      a[12 + i] = (__bf16)f3[i];
    }
#pragma unroll
    for (int nt = 0; nt < NT; ++nt) {
      const __bf16* bp = &Wl[(nt * 16 + r) * KP + k0 + kbB];
      v8bf blo = *(const v8bf*)bp;             // ds_load_b128
      v8bf bhi = *(const v8bf*)(bp + 16);      // ds_load_b128
      v16bf b = __builtin_shufflevector(blo, bhi,
                  0, 1, 2, 3, 4, 5, 6, 7, 8, 9, 10, 11, 12, 13, 14, 15);
      acc[nt] = __builtin_amdgcn_wmma_f32_16x16x32_bf16(
          /*neg_a=*/false, a, /*neg_b=*/false, b,
          /*c_mod=*/(short)0, acc[nt], /*reuse_a=*/false, /*reuse_b=*/false);
    }
  }

  const int rowBase = m0 + (hi ? 8 : 0);       // C/D: VGPR v -> row v / v+8
#pragma unroll
  for (int nt = 0; nt < NT; ++nt) {
    const int col = nt * 16 + r;
    const float bs = bias[col];
#pragma unroll
    for (int v = 0; v < 8; ++v) {
      float val = acc[nt][v] + bs;
      if (relu) val = fmaxf(val, 0.0f);
      Out[(size_t)(rowBase + v) * Nc + col] = val;
    }
  }
}

// ---------------------------------------------------------------------------
// score: one wave32 per node; per-node L2 errors + weighted score.
// out[0..N)=score, [N..2N)=attr_err, [2N..3N)=neigh_err
// ---------------------------------------------------------------------------
__global__ void score_kernel(const float* __restrict__ x,
                             const float* __restrict__ x_hat,
                             const float* __restrict__ m,
                             const float* __restrict__ m_hat,
                             float* __restrict__ out, int N) {
  const int wave = (int)(((long long)blockIdx.x * blockDim.x + threadIdx.x) >> 5);
  const int lane = (int)(threadIdx.x & 31);
  if (wave >= N) return;                       // wave-uniform
  const size_t base = (size_t)wave * FDIM;

  float ae = 0.0f, ne = 0.0f, he = 0.0f;
#pragma unroll
  for (int j = 0; j < FDIM; j += 32) {
    int i = j + lane;
    float xv = x[base + i];
    float xh = x_hat[base + i];
    float mv = m[base + i];
    float mh = m_hat[base + i];
    float d1 = xh - xv; ae += d1 * d1;
    float d2 = mh - mv; ne += d2 * d2;
    float d3 = mh - xv; he += d3 * d3;
  }
#pragma unroll
  for (int off = 16; off > 0; off >>= 1) {     // wave32 butterfly
    ae += __shfl_xor(ae, off, 32);
    ne += __shfl_xor(ne, off, 32);
    he += __shfl_xor(he, off, 32);
  }
  if (lane == 0) {
    ae = sqrtf(ae + 1e-12f);
    ne = sqrtf(ne + 1e-12f);
    he = sqrtf(he + 1e-12f);
    out[wave]                 = 1.0f * ae + 0.5f * ne + 0.3f * he;  // c,a,b
    out[(size_t)N + wave]     = ae;
    out[(size_t)2 * N + wave] = ne;
  }
}

// ---------------------------------------------------------------------------
// launch
// ---------------------------------------------------------------------------
extern "C" void kernel_launch(void* const* d_in, const int* in_sizes, int n_in,
                              void* d_out, int out_size, void* d_ws, size_t ws_size,
                              hipStream_t stream) {
  const float* x     = (const float*)d_in[0];
  const int*   eidx  = (const int*)d_in[1];
  const float* W_enc = (const float*)d_in[2];
  const float* b_enc = (const float*)d_in[3];
  const float* Wx1   = (const float*)d_in[4];
  const float* bx1   = (const float*)d_in[5];
  const float* Wx2   = (const float*)d_in[6];
  const float* bx2   = (const float*)d_in[7];
  const float* Wh1   = (const float*)d_in[8];
  const float* bh1   = (const float*)d_in[9];
  const float* Wh2   = (const float*)d_in[10];
  const float* bh2   = (const float*)d_in[11];
  float* out = (float*)d_out;

  const int N = in_sizes[0] / FDIM;      // 100000
  const int E = in_sizes[1] / 2;         // 800000
  const int* src = eidx;                 // edge_index[0]
  const int* dst = eidx + E;             // edge_index[1]

  // workspace partition (floats): deg | m | h | hm | t | x_hat | m_hat
  float* ws    = (float*)d_ws;
  float* deg   = ws;
  float* m     = deg   + (size_t)N;
  float* h     = m     + (size_t)N * FDIM;
  float* hm    = h     + (size_t)N * FDIM;
  float* t     = hm    + (size_t)N * FDIM;
  float* x_hat = t     + (size_t)N * DECD;
  float* m_hat = x_hat + (size_t)N * FDIM;

  const int BLK = 256;                   // 8 wave32s / block
  const long long NF = (long long)N * FDIM;
  const long long EF = (long long)E * FDIM;
  const unsigned gemmBlocks = (unsigned)(((N >> 4) + 7) / 8);

  // 0) init deg=1 (self loop), zero accumulators
  init_kernel<<<(unsigned)((NF + BLK - 1) / BLK), BLK, 0, stream>>>(deg, m, hm, N);
  // 1) degrees from edges
  degree_kernel<<<(unsigned)((E + BLK - 1) / BLK), BLK, 0, stream>>>(dst, deg, E);
  // 2) m_acc = segment_sum(x[src] -> dst)
  scatter_add_kernel<<<(unsigned)((EF + BLK - 1) / BLK), BLK, 0, stream>>>(x, src, dst, m, E);
  // 3) m = (m_acc + x) / deg
  finish_mean_kernel<<<(unsigned)((NF + BLK - 1) / BLK), BLK, 0, stream>>>(x, m, deg, N);
  // 4) h = relu(m @ W_enc + b_enc)                       [K=64, Nc=64]
  gemm_bias_act_kernel<64, 4><<<gemmBlocks, BLK, 0, stream>>>(m, W_enc, b_enc, h, N, 1);
  // 5) hm_acc = segment_sum(h[src] -> dst); hm = (hm_acc + h)/deg
  scatter_add_kernel<<<(unsigned)((EF + BLK - 1) / BLK), BLK, 0, stream>>>(h, src, dst, hm, E);
  finish_mean_kernel<<<(unsigned)((NF + BLK - 1) / BLK), BLK, 0, stream>>>(h, hm, deg, N);
  // 6) x_hat = relu(h @ Wx1 + bx1) @ Wx2 + bx2           [64->128->64]
  gemm_bias_act_kernel<64, 8><<<gemmBlocks, BLK, 0, stream>>>(h, Wx1, bx1, t, N, 1);
  gemm_bias_act_kernel<128, 4><<<gemmBlocks, BLK, 0, stream>>>(t, Wx2, bx2, x_hat, N, 0);
  // 7) m_hat = relu(hm @ Wh1 + bh1) @ Wh2 + bh2          [t reused, stream-ordered]
  gemm_bias_act_kernel<64, 8><<<gemmBlocks, BLK, 0, stream>>>(hm, Wh1, bh1, t, N, 1);
  gemm_bias_act_kernel<128, 4><<<gemmBlocks, BLK, 0, stream>>>(t, Wh2, bh2, m_hat, N, 0);
  // 8) per-node errors + score
  long long scoreThreads = (long long)N * 32;
  score_kernel<<<(unsigned)((scoreThreads + BLK - 1) / BLK), BLK, 0, stream>>>(
      x, x_hat, m, m_hat, out, N);
}
